// Attenion_5016521802199
// MI455X (gfx1250) — compile-verified, matrix operand order checked by reading, not run on hardware
//
#include <hip/hip_runtime.h>
#include <hip/hip_bf16.h>

typedef __attribute__((ext_vector_type(16))) _Float16 v16h;
typedef __attribute__((ext_vector_type(8)))  float    v8f;

#define SEQ  512     // B*NVARS
#define T    128
#define C    128
#define NH   8
#define HD   16
#define DFF  256
#define NROW (SEQ*T) // 65536 merged rows

// 16-bit A/B operand K mapping (CDNA5 ISA, 16-bit A 16x32):
// lane half (lane>>4) selects +8; element e of v16h -> K = 2*(e&8) + 8*half + (e&7)
__device__ __forceinline__ int kmapf(int e, int half) {
  return ((e & 8) << 1) + (half << 3) + (e & 7);
}

__device__ __forceinline__ v8f vzero() {
  v8f z = {0.f, 0.f, 0.f, 0.f, 0.f, 0.f, 0.f, 0.f};
  return z;
}

__device__ __forceinline__ v8f wmma16(v16h a, v16h b, v8f c) {
  return __builtin_amdgcn_wmma_f32_16x16x32_f16(false, a, false, b, (short)0, c, false, false);
}

// ---------------------------------------------------------------- utilities
__global__ void cvt_f32_f16(const float* __restrict__ src, _Float16* __restrict__ dst, int n) {
  int i = blockIdx.x * blockDim.x + threadIdx.x;
  if (i < n) dst[i] = (_Float16)src[i];
}

// ---------------------------------------------------------------- QKV GEMM
// rows = s*T+t (65536), cols = 384 (comp*128 + h*16 + d), K = 128
__global__ void qkv_kernel(const float* __restrict__ x, const float* __restrict__ w,
                           const float* __restrict__ bias,
                           _Float16* __restrict__ qh, _Float16* __restrict__ kh,
                           _Float16* __restrict__ vh) {
  const int mt = blockIdx.x;       // 4096 row tiles
  const int nt = blockIdx.y;       // 24 col tiles
  const int lane = threadIdx.x & 31;
  const int half = lane >> 4, mn = lane & 15;
  v8f acc = vzero();
  for (int kb = 0; kb < 128; kb += 32) {
    v16h a, b;
#pragma unroll
    for (int e = 0; e < 16; ++e) {
      int k = kmapf(e, half);
      a[e] = (_Float16)x[(size_t)(mt * 16 + mn) * C + kb + k];
      b[e] = (_Float16)w[(size_t)(kb + k) * 384 + nt * 16 + mn];
    }
    acc = wmma16(a, b, acc);
  }
#pragma unroll
  for (int r = 0; r < 8; ++r) {
    int row = mt * 16 + r + 8 * half;
    int col = nt * 16 + mn;
    float val = acc[r] + bias[col];
    int comp = col >> 7;
    int h = (col >> 4) & 7;
    int d = col & 15;
    int s = row >> 7;
    int t = row & 127;
    _Float16* dst = (comp == 0) ? qh : (comp == 1) ? kh : vh;
    dst[(((size_t)s * NH + h) * T + t) * HD + d] = (_Float16)val;
  }
}

// ---------------------------------------------------------------- EMA GEMM: dst = E @ src per (s,h)
__global__ void ema_kernel(const _Float16* __restrict__ Eh, const _Float16* __restrict__ src,
                           _Float16* __restrict__ dst) {
  const int mt = blockIdx.x & 7;   // 8 row tiles of T
  const int sh = blockIdx.x >> 3;  // s*8+h
  const int lane = threadIdx.x & 31;
  const int half = lane >> 4, mn = lane & 15;
  const _Float16* sp = src + (size_t)sh * T * HD;
  v8f acc = vzero();
  for (int kb = 0; kb < 128; kb += 32) {
    v16h a, b;
#pragma unroll
    for (int e = 0; e < 16; ++e) {
      int k = kmapf(e, half);
      a[e] = Eh[(mt * 16 + mn) * T + kb + k];
      b[e] = sp[(kb + k) * HD + mn];
    }
    acc = wmma16(a, b, acc);
  }
  _Float16* dp = dst + (size_t)sh * T * HD;
#pragma unroll
  for (int r = 0; r < 8; ++r)
    dp[(mt * 16 + r + 8 * half) * HD + mn] = (_Float16)acc[r];
}

// ---------------------------------------------------------------- attention per (s,h)
// merge_heads mapping (derived from the reference reshape/transpose):
//   (h,t,d) -> merged row tr=(h&1)*64+(t>>1), channel c=(t&1)*64+(h>>1)*16+d
__global__ void __launch_bounds__(256) attn_kernel(
    const _Float16* __restrict__ qh, const _Float16* __restrict__ kh,
    const _Float16* __restrict__ vh, const _Float16* __restrict__ eqh,
    const _Float16* __restrict__ ekh, float* __restrict__ oTok,
    float* __restrict__ oHid) {
  __shared__ _Float16 sQ[T * HD], sK[T * HD], sV[T * HD], sEQ[T * HD], sEK[T * HD];
  __shared__ _Float16 sA[8 * 16 * T];  // per-wave a_tok strips (16x128 each)
  __shared__ _Float16 sAH[16 * 16];    // a_hid
  const int tid = threadIdx.x;
  const int w = tid >> 5, lane = tid & 31, half = lane >> 4, mn = lane & 15;
  const int sh = blockIdx.x;
  const int h = sh & 7, s = sh >> 3;

  {  // cooperative loads: 2048 halves (4KB) each, 16B per thread
    const size_t base = (size_t)sh * T * HD;
    ((uint4*)sQ)[tid]  = ((const uint4*)(qh + base))[tid];
    ((uint4*)sK)[tid]  = ((const uint4*)(kh + base))[tid];
    ((uint4*)sV)[tid]  = ((const uint4*)(vh + base))[tid];
    ((uint4*)sEQ)[tid] = ((const uint4*)(eqh + base))[tid];
    ((uint4*)sEK)[tid] = ((const uint4*)(ekh + base))[tid];
  }
  __syncthreads();

  // ---- s_tok = (Eq)(Ek)^T * 4 ; each wave owns rows [w*16, w*16+16)
  v16h aq;
#pragma unroll
  for (int e = 0; e < 16; ++e) {
    int k = kmapf(e, half);
    aq[e] = (k < 16) ? sEQ[(w * 16 + mn) * HD + k] : (_Float16)0.0f;
  }
  v8f sc[8];
#pragma unroll
  for (int nt = 0; nt < 8; ++nt) {
    v16h b;
#pragma unroll
    for (int e = 0; e < 16; ++e) {
      int k = kmapf(e, half);
      b[e] = (k < 16) ? sEK[(nt * 16 + mn) * HD + k] : (_Float16)0.0f;
    }
    sc[nt] = wmma16(aq, b, vzero());
  }
  // row softmax over 128 cols: row r+8*half lives across the 16 lanes of this half
#pragma unroll
  for (int r = 0; r < 8; ++r) {
    float mx = -3.4e38f;
#pragma unroll
    for (int nt = 0; nt < 8; ++nt) mx = fmaxf(mx, sc[nt][r] * 4.0f);
    for (int msk = 1; msk <= 8; msk <<= 1) mx = fmaxf(mx, __shfl_xor(mx, msk, 32));
    float sum = 0.f;
#pragma unroll
    for (int nt = 0; nt < 8; ++nt) {
      float ev = __expf(sc[nt][r] * 4.0f - mx);
      sc[nt][r] = ev;
      sum += ev;
    }
    for (int msk = 1; msk <= 8; msk <<= 1) sum += __shfl_xor(sum, msk, 32);
    float inv = 1.0f / sum;
    int row = r + 8 * half;
#pragma unroll
    for (int nt = 0; nt < 8; ++nt)
      sA[(w * 16 + row) * T + nt * 16 + mn] = (_Float16)(sc[nt][r] * inv);
  }

  // ---- o_tok = a_tok @ v  (K=128)
  v8f oacc = vzero();
  for (int kb = 0; kb < 128; kb += 32) {
    v16h a, b;
#pragma unroll
    for (int e = 0; e < 16; ++e) {
      int k = kmapf(e, half);
      a[e] = sA[(w * 16 + mn) * T + kb + k];
      b[e] = sV[(kb + k) * HD + mn];
    }
    oacc = wmma16(a, b, oacc);
  }
#pragma unroll
  for (int r = 0; r < 8; ++r) {
    int t = w * 16 + r + 8 * half;
    int tr = (h & 1) * 64 + (t >> 1);
    int c = (t & 1) * 64 + (h >> 1) * 16 + mn;
    oTok[((size_t)s * T + tr) * C + c] = oacc[r];
  }

  // ---- s_hid = q^T k * sqrt(T)  (16x16, K=128), wave 0 only
  if (w == 0) {
    v8f sh8 = vzero();
    for (int kb = 0; kb < 128; kb += 32) {
      v16h a, b;
#pragma unroll
      for (int e = 0; e < 16; ++e) {
        int k = kmapf(e, half);
        a[e] = sQ[(kb + k) * HD + mn];  // A[m=e_row][k=time] = q[time][m]
        b[e] = sK[(kb + k) * HD + mn];  // B[k=time][n=f]   = k[time][n]
      }
      sh8 = wmma16(a, b, sh8);
    }
#pragma unroll
    for (int r = 0; r < 8; ++r) {
      float x = sh8[r] * 11.3137085f;  // sqrt(128)
      float mx = x;
      for (int msk = 1; msk <= 8; msk <<= 1) mx = fmaxf(mx, __shfl_xor(mx, msk, 32));
      float ev = __expf(x - mx);
      float sum = ev;
      for (int msk = 1; msk <= 8; msk <<= 1) sum += __shfl_xor(sum, msk, 32);
      sAH[(r + 8 * half) * 16 + mn] = (_Float16)(ev / sum);
    }
  }
  __syncthreads();

  // ---- o_hid = v @ a_hid^T  (128x16, K=16 padded to 32)
  v8f ohid;
  {
    v16h a, b;
#pragma unroll
    for (int e = 0; e < 16; ++e) {
      int k = kmapf(e, half);
      a[e] = (k < 16) ? sV[(w * 16 + mn) * HD + k] : (_Float16)0.0f;
      b[e] = (k < 16) ? sAH[mn * 16 + k] : (_Float16)0.0f;
    }
    ohid = wmma16(a, b, vzero());
  }
#pragma unroll
  for (int r = 0; r < 8; ++r) {
    int t = w * 16 + r + 8 * half;
    int tr = (h & 1) * 64 + (t >> 1);
    int c = (t & 1) * 64 + (h >> 1) * 16 + mn;
    oHid[((size_t)s * T + tr) * C + c] = ohid[r];
  }
}

// ---------------------------------------------------------------- batchnorm stats (deterministic 2-stage)
__global__ void stats_partial(const float* __restrict__ x, float* __restrict__ part) {
  const int c = threadIdx.x;  // 128 channels
  const int b = blockIdx.x;   // 256 blocks, 256 rows each
  float s = 0.f, s2 = 0.f;
  for (int r = 0; r < 256; ++r) {
    float v = x[((size_t)b * 256 + r) * C + c];
    s += v;
    s2 += v * v;
  }
  part[b * C + c] = s;
  part[256 * C + b * C + c] = s2;
}

__global__ void stats_reduce(const float* __restrict__ part, float* __restrict__ mv) {
  const int c = threadIdx.x;
  float s = 0.f, s2 = 0.f;
  for (int b = 0; b < 256; ++b) {
    s += part[b * C + c];
    s2 += part[256 * C + b * C + c];
  }
  float mean = s * (1.0f / 65536.0f);
  float var = s2 * (1.0f / 65536.0f) - mean * mean;
  mv[c] = mean;
  mv[C + c] = rsqrtf(var + 1e-5f);
}

// ---------------------------------------------------------------- FF layer 1: gelu(bn(x) @ w1 + b1) -> f16
__global__ void ff1_kernel(const float* __restrict__ x, const float* __restrict__ mv,
                           const float* __restrict__ g, const float* __restrict__ bta,
                           const float* __restrict__ w1, const float* __restrict__ b1,
                           _Float16* __restrict__ hout) {
  const int mt = blockIdx.x, nt = blockIdx.y;
  const int lane = threadIdx.x & 31, half = lane >> 4, mn = lane & 15;
  v8f acc = vzero();
  for (int kb = 0; kb < 128; kb += 32) {
    v16h a, b;
#pragma unroll
    for (int e = 0; e < 16; ++e) {
      int k = kb + kmapf(e, half);
      float xv = (x[(size_t)(mt * 16 + mn) * C + k] - mv[k]) * mv[C + k] * g[k] + bta[k];
      a[e] = (_Float16)xv;
      b[e] = (_Float16)w1[(size_t)k * DFF + nt * 16 + mn];
    }
    acc = wmma16(a, b, acc);
  }
  int col = nt * 16 + mn;
#pragma unroll
  for (int r = 0; r < 8; ++r) {
    int row = mt * 16 + r + 8 * half;
    float z = acc[r] + b1[col];
    float ge = 0.5f * z * (1.0f + erff(z * 0.70710678118f));  // exact gelu
    hout[(size_t)row * DFF + col] = (_Float16)ge;
  }
}

// ---------------------------------------------------------------- FF layer 2: dst = base + h @ w2 + b2
__global__ void ff2_kernel(const _Float16* __restrict__ hin, const float* __restrict__ w2,
                           const float* __restrict__ b2, const float* base, float* dst) {
  const int mt = blockIdx.x, nt = blockIdx.y;
  const int lane = threadIdx.x & 31, half = lane >> 4, mn = lane & 15;
  v8f acc = vzero();
  for (int kb = 0; kb < 256; kb += 32) {
    v16h a, b;
#pragma unroll
    for (int e = 0; e < 16; ++e) {
      int k = kb + kmapf(e, half);
      a[e] = hin[(size_t)(mt * 16 + mn) * DFF + k];
      b[e] = (_Float16)w2[(size_t)k * C + nt * 16 + mn];
    }
    acc = wmma16(a, b, acc);
  }
  int col = nt * 16 + mn;
#pragma unroll
  for (int r = 0; r < 8; ++r) {
    size_t idx = (size_t)(mt * 16 + r + 8 * half) * C + col;
    dst[idx] = base[idx] + acc[r] + b2[col];
  }
}

// ---------------------------------------------------------------- final batchnorm
__global__ void final_norm(const float* __restrict__ x, const float* __restrict__ mv,
                           const float* __restrict__ g, const float* __restrict__ b,
                           float* __restrict__ out, int n) {
  int i = blockIdx.x * blockDim.x + threadIdx.x;
  if (i < n) {
    int c = i & 127;
    out[i] = (x[i] - mv[c]) * mv[C + c] * g[c] + b[c];
  }
}

extern "C" void kernel_launch(void* const* d_in, const int* in_sizes, int n_in,
                              void* d_out, int out_size, void* d_ws, size_t ws_size,
                              hipStream_t stream) {
  (void)in_sizes; (void)n_in; (void)out_size; (void)ws_size;
  const float* src    = (const float*)d_in[0];
  const float* qkv_w  = (const float*)d_in[1];
  const float* qkv_b  = (const float*)d_in[2];
  const float* ff1_w1 = (const float*)d_in[3];
  const float* ff1_b1 = (const float*)d_in[4];
  const float* ff1_w2 = (const float*)d_in[5];
  const float* ff1_b2 = (const float*)d_in[6];
  const float* ff2_w1 = (const float*)d_in[7];
  const float* ff2_b1 = (const float*)d_in[8];
  const float* ff2_w2 = (const float*)d_in[9];
  const float* ff2_b2 = (const float*)d_in[10];
  const float* g1 = (const float*)d_in[11];
  const float* b1 = (const float*)d_in[12];
  const float* g2 = (const float*)d_in[13];
  const float* b2 = (const float*)d_in[14];
  const float* g3 = (const float*)d_in[15];
  const float* b3 = (const float*)d_in[16];
  const float* ema = (const float*)d_in[17];
  float* out = (float*)d_out;

  char* ws = (char*)d_ws;
  size_t off = 0;
  auto alloc = [&](size_t bytes) -> void* {
    void* p = ws + off;
    off += (bytes + 255) & ~(size_t)255;
    return p;
  };
  const size_t HE = (size_t)SEQ * NH * T * HD;  // 8,388,608 elems per head tensor
  _Float16* qh  = (_Float16*)alloc(HE * 2);
  _Float16* kh  = (_Float16*)alloc(HE * 2);
  _Float16* vh  = (_Float16*)alloc(HE * 2);
  _Float16* eqh = (_Float16*)alloc(HE * 2);
  _Float16* ekh = (_Float16*)alloc(HE * 2);
  _Float16* Eh  = (_Float16*)alloc((size_t)T * T * 2);
  float* oTok = (float*)alloc((size_t)NROW * C * 4);
  float* oHid = (float*)alloc((size_t)NROW * C * 4);
  _Float16* h1 = (_Float16*)alloc((size_t)NROW * DFF * 2);
  float* accY  = (float*)alloc((size_t)NROW * C * 4);
  float* part  = (float*)alloc((size_t)2 * 256 * C * 4);
  float* mv1   = (float*)alloc(2 * C * 4);
  float* mv2   = (float*)alloc(2 * C * 4);
  float* mv3   = (float*)alloc(2 * C * 4);

  cvt_f32_f16<<<(T * T + 255) / 256, 256, 0, stream>>>(ema, Eh, T * T);
  qkv_kernel<<<dim3(NROW / 16, 24), 32, 0, stream>>>(src, qkv_w, qkv_b, qh, kh, vh);
  ema_kernel<<<SEQ * NH * 8, 32, 0, stream>>>(Eh, qh, eqh);
  ema_kernel<<<SEQ * NH * 8, 32, 0, stream>>>(Eh, kh, ekh);
  attn_kernel<<<SEQ * NH, 256, 0, stream>>>(qh, kh, vh, eqh, ekh, oTok, oHid);

  stats_partial<<<256, C, 0, stream>>>(oTok, part);
  stats_reduce<<<1, C, 0, stream>>>(part, mv1);
  stats_partial<<<256, C, 0, stream>>>(oHid, part);
  stats_reduce<<<1, C, 0, stream>>>(part, mv2);

  ff1_kernel<<<dim3(NROW / 16, DFF / 16), 32, 0, stream>>>(oTok, mv1, g1, b1, ff1_w1, ff1_b1, h1);
  ff2_kernel<<<dim3(NROW / 16, C / 16), 32, 0, stream>>>(h1, ff1_w2, ff1_b2, src, accY);
  ff1_kernel<<<dim3(NROW / 16, DFF / 16), 32, 0, stream>>>(oHid, mv2, g2, b2, ff2_w1, ff2_b1, h1);
  ff2_kernel<<<dim3(NROW / 16, C / 16), 32, 0, stream>>>(h1, ff2_w2, ff2_b2, accY, accY);

  stats_partial<<<256, C, 0, stream>>>(accY, part);
  stats_reduce<<<1, C, 0, stream>>>(part, mv3);
  final_norm<<<(NROW * C + 255) / 256, 256, 0, stream>>>(accY, mv3, g3, b3, out, NROW * C);
}
